// SparseMoE_85160611545784
// MI455X (gfx1250) — compile-verified, matrix operand order checked by reading, not run on hardware
//
#include <hip/hip_runtime.h>
#include <hip/hip_bf16.h>

// ---------------------------------------------------------------------------
// Sparse MoE (SwiGLU, top-2 of 8 experts), D=1024, H=2048, fp32 in/out.
//  k0: zero out + counters
//  k1: router (wave32/token) + per-expert compaction
//  k2 (x3): weight fp32 -> bf16 hi/lo, pre-swizzled into WMMA B-frag order
//  k3: fused expert FFN, v_wmma_f32_16x16x32_bf16 with 2-term hi/lo split
// ---------------------------------------------------------------------------

#define MOE_D 1024
#define MOE_H 2048
#define MOE_E 8
#define MTILE 32     // tokens per workgroup in the expert kernel
#define HCHUNK 64    // hidden columns processed per outer iteration

typedef __attribute__((ext_vector_type(16))) __bf16 bf16x16;
typedef __attribute__((ext_vector_type(8)))  float  v8f;

union Frag {
  bf16x16 v;
  unsigned u[8];
  uint4    q[2];
};

// --- truncation-based fp32 -> bf16 hi/lo split of a pair, packed with
// v_perm_b32. lo absorbs hi's truncation error exactly; total err ~2^-16.
__device__ __forceinline__ void split2(float f0, float f1, unsigned& hp, unsigned& lp) {
  unsigned u0 = __float_as_uint(f0), u1 = __float_as_uint(f1);
  hp = __builtin_amdgcn_perm(u1, u0, 0x07060302u);   // {hi16(f1), hi16(f0)}
  float r0 = f0 - __uint_as_float(u0 & 0xffff0000u);
  float r1 = f1 - __uint_as_float(u1 & 0xffff0000u);
  lp = __builtin_amdgcn_perm(__float_as_uint(r1), __float_as_uint(r0), 0x07060302u);
}

// --- A-matrix fragment from LDS (row-major bf16, documented 16-bit A layout):
// lane<16: M=lane, K in {k..k+7, k+16..k+23}; lane>=16: M=lane-16, K offset +8.
__device__ __forceinline__ Frag load_a_frag(const __bf16* row, int koff) {
  Frag f;
  f.q[0] = *reinterpret_cast<const uint4*>(row + koff);
  f.q[1] = *reinterpret_cast<const uint4*>(row + koff + 16);
  return f;
}

// --- B-fragment from pre-swizzled bf16 weights: lane's 16 elements are
// contiguous (32 B) -> two global_load_b128, fully coalesced per wave.
__device__ __forceinline__ Frag load_b_frag_swz(const __bf16* blk, int lane) {
  Frag f;
  const uint4* p = reinterpret_cast<const uint4*>(blk + lane * 16);
  f.q[0] = p[0];
  f.q[1] = p[1];
  return f;
}

// --- fallback: B-fragment straight from global fp32 weights, split inline.
__device__ __forceinline__ void load_b_frag_f32(const float* __restrict__ base, int ld,
                                                int koff, Frag& bh, Frag& bl) {
#pragma unroll
  for (int j = 0; j < 8; ++j) {
    int kk = ((j & 4) << 2) + 2 * (j & 3) + koff;   // (0,2,4,6,16,18,20,22)+koff
    float f0 = base[(size_t)kk * ld];
    float f1 = base[(size_t)(kk + 1) * ld];
    split2(f0, f1, bh.u[j], bl.u[j]);
  }
}

#define WMMA_BF16(A, B, C) \
  __builtin_amdgcn_wmma_f32_16x16x32_bf16(false, (A), false, (B), (short)0, (C), false, false)

// ---------------------------------------------------------------------------
// Kernel 0: zero output + expert counters (graph replay does not re-poison;
// the exactly-2-commutative-atomic-adds determinism needs out == 0).
// ---------------------------------------------------------------------------
__global__ void moe_zero_kernel(float* __restrict__ out, int n, int* __restrict__ cnt) {
  int i = blockIdx.x * blockDim.x + threadIdx.x;
  if (i < n) out[i] = 0.0f;
  if (blockIdx.x == 0 && threadIdx.x < MOE_E) cnt[threadIdx.x] = 0;
}

// ---------------------------------------------------------------------------
// Kernel 1: router. One wave32 per token, shuffle reduction, top-2 with
// lowest-index tie-break (matches jax top_k), softmax, atomic compaction.
// ---------------------------------------------------------------------------
__global__ void moe_router_kernel(const float* __restrict__ x,
                                  const float* __restrict__ rw,
                                  int* __restrict__ tokList,
                                  float* __restrict__ wList,
                                  int* __restrict__ cnt, int T) {
  int wave = (blockIdx.x * blockDim.x + threadIdx.x) >> 5;
  int lane = threadIdx.x & 31;
  if (wave >= T) return;
  const float* xr = x + (size_t)wave * MOE_D;

  float a[MOE_E];
#pragma unroll
  for (int e = 0; e < MOE_E; ++e) a[e] = 0.0f;

  for (int d = lane; d < MOE_D; d += 32) {
    float xv = xr[d];
#pragma unroll
    for (int e = 0; e < MOE_E; ++e) a[e] = fmaf(xv, rw[d * MOE_E + e], a[e]);
  }
#pragma unroll
  for (int off = 16; off > 0; off >>= 1) {
#pragma unroll
    for (int e = 0; e < MOE_E; ++e) a[e] += __shfl_down(a[e], off, 32);
  }

  if (lane == 0) {
    int i0 = 0; float l0 = a[0];
#pragma unroll
    for (int e = 1; e < MOE_E; ++e) if (a[e] > l0) { l0 = a[e]; i0 = e; }
    int i1 = -1; float l1 = -3.4e38f;
#pragma unroll
    for (int e = 0; e < MOE_E; ++e) if (e != i0 && a[e] > l1) { l1 = a[e]; i1 = e; }
    float e1 = __expf(l1 - l0);
    float inv = 1.0f / (1.0f + e1);
    float w0 = inv, w1 = e1 * inv;

    int p0 = atomicAdd(&cnt[i0], 1);
    tokList[i0 * T + p0] = wave;  wList[i0 * T + p0] = w0;
    int p1 = atomicAdd(&cnt[i1], 1);
    tokList[i1 * T + p1] = wave;  wList[i1 * T + p1] = w1;
  }
}

// ---------------------------------------------------------------------------
// Kernel 2: convert one fp32 weight matrix (E x Kdim x Ndim, row-major) into
// bf16 hi/lo, stored in WMMA B-fragment order:
//   element offset = (((e*KB + kb)*NG + ng)*32 + lane)*16 + j*2
// so the GEMM loads each fragment as two b128s. One thread per packed pair.
// ---------------------------------------------------------------------------
__global__ void moe_swizzle_kernel(const float* __restrict__ W,
                                   __bf16* __restrict__ hiOut,
                                   __bf16* __restrict__ loOut,
                                   int Kdim, int Ndim) {
  size_t idx = (size_t)blockIdx.x * blockDim.x + threadIdx.x;
  size_t total = (size_t)MOE_E * Kdim * Ndim / 2;
  if (idx >= total) return;

  int j    = (int)(idx & 7);
  int lane = (int)((idx >> 3) & 31);
  size_t rest = idx >> 8;
  int NG = Ndim / 16;
  int ng = (int)(rest % NG); rest /= NG;
  int KB = Kdim / 32;
  int kb = (int)(rest % KB);
  int e  = (int)(rest / KB);

  int ln  = lane & 15;
  int lhi = (lane >> 4) ? 8 : 0;
  int kk  = ((j & 4) << 2) + 2 * (j & 3) + lhi;

  size_t src = ((size_t)e * Kdim + (size_t)kb * 32 + kk) * Ndim + (size_t)ng * 16 + ln;
  float f0 = W[src];
  float f1 = W[src + Ndim];
  unsigned hp, lp;
  split2(f0, f1, hp, lp);
  *reinterpret_cast<unsigned*>(hiOut + idx * 2) = hp;
  *reinterpret_cast<unsigned*>(loOut + idx * 2) = lp;
}

// ---------------------------------------------------------------------------
// Kernel 3: fused expert FFN over compacted tokens.
// grid = (T/MTILE, E), 256 threads (8 waves).
// ---------------------------------------------------------------------------
template <bool PRE>
__global__ __launch_bounds__(256)
void moe_expert_kernel(const float* __restrict__ x,
                       const float* __restrict__ w1,
                       const float* __restrict__ wg,
                       const float* __restrict__ w2,
                       const __bf16* __restrict__ w1h, const __bf16* __restrict__ w1l,
                       const __bf16* __restrict__ wgh, const __bf16* __restrict__ wgl,
                       const __bf16* __restrict__ w2h, const __bf16* __restrict__ w2l,
                       const int* __restrict__ tokList,
                       const float* __restrict__ wList,
                       const int* __restrict__ cnt,
                       float* __restrict__ out, int T) {
  const int e    = blockIdx.y;
  const int tile = blockIdx.x;
  const int nTok = cnt[e];
  const int start = tile * MTILE;
  if (start >= nTok) return;   // uniform across block: safe before barriers

  extern __shared__ char smem[];
  __bf16* Ahi  = (__bf16*)smem;                       // [MTILE][D]
  __bf16* Alo  = Ahi + MTILE * MOE_D;                 // [MTILE][D]
  __bf16* hHi  = Alo + MTILE * MOE_D;                 // [MTILE][HCHUNK]
  __bf16* hLo  = hHi + MTILE * HCHUNK;                // [MTILE][HCHUNK]
  int*    tTok = (int*)(hLo + MTILE * HCHUNK);        // [MTILE]
  float*  tWgt = (float*)(tTok + MTILE);              // [MTILE]

  const int wv   = threadIdx.x >> 5;       // wave 0..7
  const int lane = threadIdx.x & 31;
  const int ln   = lane & 15;
  const int lhi  = (lane >> 4) ? 8 : 0;

  // ---- token ids + combine weights ---------------------------------------
  if (threadIdx.x < MTILE) {
    int i = start + (int)threadIdx.x;
    int tok; float wgt;
    if (i < nTok) { tok = tokList[e * T + i];     wgt = wList[e * T + i]; }
    else          { tok = tokList[e * T + start]; wgt = 0.0f; }   // pad
    tTok[threadIdx.x] = tok;
    tWgt[threadIdx.x] = wgt;
  }
  __syncthreads();

  // ---- gather x rows -> LDS bf16 hi/lo (coalesced float4 loads) -----------
  for (int idx = threadIdx.x; idx < MTILE * (MOE_D / 4); idx += 256) {
    int m  = idx >> 8;        // MOE_D/4 == 256
    int c4 = idx & 255;
    const float4 f = reinterpret_cast<const float4*>(x + (size_t)tTok[m] * MOE_D)[c4];
    uint2 ph, pl;
    split2(f.x, f.y, ph.x, pl.x);
    split2(f.z, f.w, ph.y, pl.y);
    *reinterpret_cast<uint2*>(Ahi + m * MOE_D + c4 * 4) = ph;
    *reinterpret_cast<uint2*>(Alo + m * MOE_D + c4 * 4) = pl;
  }
  __syncthreads();

  const float*  w1e  = w1 + (size_t)e * MOE_D * MOE_H;
  const float*  wge  = wg + (size_t)e * MOE_D * MOE_H;
  const float*  w2e  = w2 + (size_t)e * MOE_H * MOE_D;
  const size_t  eOff = (size_t)e * MOE_D * MOE_H;     // elements per expert

  const int mi = wv & 1;      // phase-1 token-row tile
  const int ni = wv >> 1;     // phase-1 h-col tile

  v8f acc[2][8];
#pragma unroll
  for (int mt = 0; mt < 2; ++mt)
#pragma unroll
    for (int nt = 0; nt < 8; ++nt) acc[mt][nt] = {};

  const __bf16* aRowHi1 = Ahi + (size_t)(mi * 16 + ln) * MOE_D;
  const __bf16* aRowLo1 = Alo + (size_t)(mi * 16 + ln) * MOE_D;

  for (int hc = 0; hc < MOE_H; hc += HCHUNK) {
    // ---------------- phase 1: h tile = (x@w1) * silu(x@wg) ----------------
    v8f c1 = {}, cg = {};
#pragma unroll 2
    for (int k0 = 0; k0 < MOE_D; k0 += 32) {
      Frag ah = load_a_frag(aRowHi1, k0 + lhi);
      Frag al = load_a_frag(aRowLo1, k0 + lhi);
      Frag b1h, b1l, bgh, bgl;
      if (PRE) {
        // block index: (kb * (H/16) + (hc/16 + ni)) * 512 elements
        size_t blk = eOff + ((size_t)(k0 >> 5) * (MOE_H / 16) + ((hc >> 4) + ni)) * 512;
        b1h = load_b_frag_swz(w1h + blk, lane);
        b1l = load_b_frag_swz(w1l + blk, lane);
        bgh = load_b_frag_swz(wgh + blk, lane);
        bgl = load_b_frag_swz(wgl + blk, lane);
      } else {
        const int hcol = hc + ni * 16 + ln;
        load_b_frag_f32(w1e + (size_t)k0 * MOE_H + hcol, MOE_H, lhi, b1h, b1l);
        load_b_frag_f32(wge + (size_t)k0 * MOE_H + hcol, MOE_H, lhi, bgh, bgl);
      }
      c1 = WMMA_BF16(ah.v, b1h.v, c1);
      c1 = WMMA_BF16(ah.v, b1l.v, c1);
      c1 = WMMA_BF16(al.v, b1h.v, c1);
      cg = WMMA_BF16(ah.v, bgh.v, cg);
      cg = WMMA_BF16(ah.v, bgl.v, cg);
      cg = WMMA_BF16(al.v, bgh.v, cg);
    }

    __syncthreads();   // prior phase-2 reads of hHi/hLo are done
#pragma unroll
    for (int j = 0; j < 8; ++j) {
      float g  = cg[j];
      float hv = c1[j] * (g / (1.0f + __expf(-g)));
      unsigned uh = __float_as_uint(hv);
      float lo = hv - __uint_as_float(uh & 0xffff0000u);
      int m   = mi * 16 + j + lhi;
      int col = ni * 16 + ln;
      unsigned short hs = (unsigned short)(uh >> 16);
      unsigned short ls = (unsigned short)(__float_as_uint(lo) >> 16);
      hHi[m * HCHUNK + col] = *reinterpret_cast<__bf16*>(&hs);
      hLo[m * HCHUNK + col] = *reinterpret_cast<__bf16*>(&ls);
    }
    __syncthreads();   // h chunk visible to all waves

    // ---------------- phase 2: acc += h_chunk @ w2[hc:hc+64, :] ------------
#pragma unroll
    for (int ks = 0; ks < HCHUNK / 32; ++ks) {
      const int k0 = ks * 32;
      Frag ah[2], al[2];
#pragma unroll
      for (int mt = 0; mt < 2; ++mt) {
        ah[mt] = load_a_frag(hHi + (size_t)(mt * 16 + ln) * HCHUNK, k0 + lhi);
        al[mt] = load_a_frag(hLo + (size_t)(mt * 16 + ln) * HCHUNK, k0 + lhi);
      }
#pragma unroll
      for (int nt = 0; nt < 8; ++nt) {
        Frag bh, bl;
        if (PRE) {
          size_t blk = eOff + ((size_t)((hc + k0) >> 5) * (MOE_D / 16) + (wv * 8 + nt)) * 512;
          bh = load_b_frag_swz(w2h + blk, lane);
          bl = load_b_frag_swz(w2l + blk, lane);
        } else {
          const int col = wv * 128 + nt * 16 + ln;
          load_b_frag_f32(w2e + (size_t)(hc + k0) * MOE_D + col, MOE_D, lhi, bh, bl);
        }
#pragma unroll
        for (int mt = 0; mt < 2; ++mt) {
          acc[mt][nt] = WMMA_BF16(ah[mt].v, bh.v, acc[mt][nt]);
          acc[mt][nt] = WMMA_BF16(ah[mt].v, bl.v, acc[mt][nt]);
          acc[mt][nt] = WMMA_BF16(al[mt].v, bh.v, acc[mt][nt]);
        }
      }
    }
  }

  // ---- combine: out[token] += w_combine * acc  (2 atomic adds / element) --
#pragma unroll
  for (int mt = 0; mt < 2; ++mt) {
#pragma unroll
    for (int j = 0; j < 8; ++j) {
      int m = mt * 16 + j + lhi;
      float wgt = tWgt[m];
      if (wgt != 0.0f) {
        size_t rowOff = (size_t)tTok[m] * MOE_D;
#pragma unroll
        for (int nt = 0; nt < 8; ++nt) {
          int col = wv * 128 + nt * 16 + ln;
          atomicAdd(out + rowOff + col, wgt * acc[mt][nt][j]);
        }
      }
    }
  }
}

// ---------------------------------------------------------------------------
extern "C" void kernel_launch(void* const* d_in, const int* in_sizes, int n_in,
                              void* d_out, int out_size, void* d_ws, size_t ws_size,
                              hipStream_t stream) {
  const float* x  = (const float*)d_in[0];
  const float* rw = (const float*)d_in[1];
  const float* w1 = (const float*)d_in[2];
  const float* wg = (const float*)d_in[3];
  const float* w2 = (const float*)d_in[4];
  float* out = (float*)d_out;

  const int T = in_sizes[0] / MOE_D;   // 8192

  // --- workspace layout ----------------------------------------------------
  // [tokList E*T ints][wList E*T floats][cnt E ints][pad to 256]
  // [w1h][w1l][wgh][wgl][w2h][w2l] each E*D*H bf16 (pre-swizzled), if room.
  char* ws = (char*)d_ws;
  int*   tokList = (int*)ws;
  float* wListP  = (float*)(ws + sizeof(int) * (size_t)MOE_E * T);
  int*   cntP    = (int*)(ws + (sizeof(int) + sizeof(float)) * (size_t)MOE_E * T);
  size_t listBytes = ((sizeof(int) + sizeof(float)) * (size_t)MOE_E * T
                      + sizeof(int) * MOE_E + 255) & ~(size_t)255;
  const size_t wElems = (size_t)MOE_E * MOE_D * MOE_H;   // per hi/lo array
  const size_t wBytes = wElems * sizeof(__bf16);
  const bool pre = ws_size >= listBytes + 6 * wBytes;

  __bf16* w1h = (__bf16*)(ws + listBytes);
  __bf16* w1l = w1h + wElems;
  __bf16* wgh = w1l + wElems;
  __bf16* wgl = wgh + wElems;
  __bf16* w2h = wgl + wElems;
  __bf16* w2l = w2h + wElems;

  const int nOut = T * MOE_D;
  moe_zero_kernel<<<(nOut + 255) / 256, 256, 0, stream>>>(out, nOut, cntP);

  moe_router_kernel<<<(T * 32) / 256, 256, 0, stream>>>(x, rw, tokList, wListP, cntP, T);

  if (pre) {
    const size_t pairs = wElems / 2;                    // threads per matrix
    const int blks = (int)((pairs + 255) / 256);
    moe_swizzle_kernel<<<blks, 256, 0, stream>>>(w1, w1h, w1l, MOE_D, MOE_H);
    moe_swizzle_kernel<<<blks, 256, 0, stream>>>(wg, wgh, wgl, MOE_D, MOE_H);
    moe_swizzle_kernel<<<blks, 256, 0, stream>>>(w2, w2h, w2l, MOE_H, MOE_D);
  }

  dim3 grid(T / MTILE, MOE_E);
  size_t smemBytes = (size_t)(2 * MTILE * MOE_D + 2 * MTILE * HCHUNK) * sizeof(__bf16)
                   + (size_t)MTILE * (sizeof(int) + sizeof(float));
  if (pre) {
    moe_expert_kernel<true><<<grid, 256, smemBytes, stream>>>(
        x, w1, wg, w2, w1h, w1l, wgh, wgl, w2h, w2l,
        tokList, wListP, cntP, out, T);
  } else {
    moe_expert_kernel<false><<<grid, 256, smemBytes, stream>>>(
        x, w1, wg, w2, nullptr, nullptr, nullptr, nullptr, nullptr, nullptr,
        tokList, wListP, cntP, out, T);
  }
}
